// FlashAttentionV3_27685359190823
// MI455X (gfx1250) — matrix operand order repaired and checked by reading, not compiled, not run on hardware
//
#include <hip/hip_runtime.h>
#include <hip/hip_bf16.h>

typedef __attribute__((ext_vector_type(16))) __bf16 v16bf;
typedef __attribute__((ext_vector_type(8)))  __bf16 v8bf;
typedef __attribute__((ext_vector_type(8)))  float  v8f;
typedef __attribute__((ext_vector_type(4)))  unsigned int u32x4;
typedef __attribute__((ext_vector_type(8)))  int i32x8;
typedef __attribute__((ext_vector_type(4)))  int i32x4;

constexpr int Bsz = 2;
constexpr int S   = 2048;
constexpr int D   = 1024;
constexpr int H   = 16;
constexpr int HD  = 64;
constexpr int M   = Bsz * S;            // 4096 rows
// log2(10000)/32
constexpr float ROPE_L2 = 0.41524101186605623f;

#if __has_include(<hip/amd_detail/amd_gfx1250_TDM.h>)
#define TDM_SIX_ARGS 1
#else
#define TDM_SIX_ARGS 0
#endif

union BF16x16 { v16bf v; v8bf h[2]; };

// ---- WMMA fragment loaders (row-major source, NT gemm) -------------------
// A operand 16x32 bf16: lane<16 -> row=lane, K {k0..k0+7} and {k0+16..k0+23}
//                       lane>=16 -> row=lane-16, K {k0+8..} and {k0+24..}
__device__ __forceinline__ v16bf load_frag_a(const __bf16* base, int ld, int k0, int lane) {
    int mr = lane & 15;
    int kh = (lane >> 4) << 3;              // 0 or 8
    const __bf16* p = base + mr * ld + k0 + kh;
    BF16x16 f;
    f.h[0] = *(const v8bf*)(p);
    f.h[1] = *(const v8bf*)(p + 16);
    return f.v;
}
// B operand 32x16 bf16 from row-major [N][K] storage:
//   lane<16 -> col N=lane, K {k0..k0+7} and {k0+16..k0+23}; lane>=16 -> K {k0+8..} and {k0+24..}
__device__ __forceinline__ v16bf load_frag_b(const __bf16* base, int ld, int k0, int lane) {
    int nr = lane & 15;
    int kh = (lane >> 4) << 3;              // 0 or 8
    const __bf16* p = base + nr * ld + k0 + kh;
    BF16x16 f;
    f.h[0] = *(const v8bf*)(p);
    f.h[1] = *(const v8bf*)(p + 16);
    return f.v;
}

__device__ __forceinline__ v8f wmma_bf16(v16bf a, v16bf b, v8f c) {
    return __builtin_amdgcn_wmma_f32_16x16x32_bf16(false, a, false, b, (short)0, c, false, false);
}

__device__ __forceinline__ float redmax16(float v) {
    v = fmaxf(v, __shfl_xor(v, 8, 16));
    v = fmaxf(v, __shfl_xor(v, 4, 16));
    v = fmaxf(v, __shfl_xor(v, 2, 16));
    v = fmaxf(v, __shfl_xor(v, 1, 16));
    return v;
}
__device__ __forceinline__ float redsum16(float v) {
    v += __shfl_xor(v, 8, 16);
    v += __shfl_xor(v, 4, 16);
    v += __shfl_xor(v, 2, 16);
    v += __shfl_xor(v, 1, 16);
    return v;
}

// ---- Tensor Data Mover: 2D tile (bf16) global -> LDS ----------------------
// D# per CDNA5 ISA 8.3/8.4: group0 = {count,lds_addr,global_addr,type=2},
// group1 = {data_size=2B, tensor dims, 48b stride0, tile dims}. 2D only
// (groups 2/3 zero). Tracked by TENSORcnt.
__device__ __forceinline__ unsigned lds_off(const void* p) {
    return (unsigned)(unsigned long long)(size_t)p;   // low 32b of LDS aperture addr
}
__device__ __forceinline__ void tdm_load_2d(unsigned ldsAddr, unsigned long long gaddr,
                                            unsigned tensor_d0, unsigned tensor_d1,
                                            unsigned long long stride0,
                                            unsigned tile_d0, unsigned tile_d1) {
    u32x4 g0;
    g0[0] = 1u;                                              // count=1, user mode
    g0[1] = ldsAddr;                                         // lds_addr (bytes)
    g0[2] = (unsigned)gaddr;                                 // global_addr[31:0]
    g0[3] = (unsigned)((gaddr >> 32) & 0x1FFFFFFull) | (2u << 30);  // [56:32] | type=2
    i32x8 g1;
    g1[0] = (int)(1u << 16);                                 // data_size=1 -> 2 bytes
    g1[1] = (int)((tensor_d0 & 0xFFFFu) << 16);              // tensor_dim0[15:0] @ bits 63:48
    g1[2] = (int)(((tensor_d0 >> 16) & 0xFFFFu) | ((tensor_d1 & 0xFFFFu) << 16));
    g1[3] = (int)(((tensor_d1 >> 16) & 0xFFFFu) | ((tile_d0 & 0xFFFFu) << 16));
    g1[4] = (int)(tile_d1 & 0xFFFFu);                        // tile_dim1; tile_dim2=0
    g1[5] = (int)(unsigned)stride0;                          // tensor_dim0_stride[31:0]
    g1[6] = (int)((unsigned)(stride0 >> 32) & 0xFFFFu);      // stride0[47:32]; dim1_stride lo=0
    g1[7] = 0;
    i32x4 z4 = {0, 0, 0, 0};
#if TDM_SIX_ARGS
    i32x8 z8 = {0, 0, 0, 0, 0, 0, 0, 0};
    __builtin_amdgcn_tensor_load_to_lds(g0, g1, z4, z4, z8, 0);
#else
    __builtin_amdgcn_tensor_load_to_lds(g0, g1, z4, z4, 0);
#endif
}

// ---- fp32 -> bf16 convert -------------------------------------------------
__global__ void cvt_bf16_kernel(const float* __restrict__ src, __bf16* __restrict__ dst, int n) {
    int i = blockIdx.x * blockDim.x + threadIdx.x;
    if (i < n) dst[i] = (__bf16)src[i];
}

// ---- NT GEMM: C[m,n] = sum_k A[m,k]*W[n,k] --------------------------------
// MODE 0: qkv projection. Epilogue: RoPE on q/k, q*=1/8; store bf16 to
//         Qo/Ko as [B][H][S][64], V transposed to Vto as [B][H][64][S].
// MODE 1: out projection. Epilogue: + bias, store fp32 to outF [M][1024].
// Block: 256 threads = 8 waves; block tile 128(M) x 64(N); K step 32.
template <int MODE>
__global__ void __launch_bounds__(256)
gemm_nt_kernel(const __bf16* __restrict__ A, const __bf16* __restrict__ W,
               float* __restrict__ outF, const float* __restrict__ bias,
               __bf16* __restrict__ Qo, __bf16* __restrict__ Ko, __bf16* __restrict__ Vto) {
    __shared__ __bf16 Bt[2][64 * 32];       // double-buffered W tile (8 KB)
    const int wave = threadIdx.x >> 5;
    const int lane = threadIdx.x & 31;
    const int half = lane >> 4;
    const int nr   = lane & 15;
    const int m0w   = blockIdx.x * 128 + wave * 16;
    const int nBase = blockIdx.y * 64;
    const __bf16* Arow = A + m0w * D;

    v8f acc[4] = {};

    const int row = threadIdx.x >> 2;       // 0..63
    const int cc  = (threadIdx.x & 3) << 3; // 0,8,16,24
    *(v8bf*)&Bt[0][row * 32 + cc] = *(const v8bf*)&W[(nBase + row) * D + cc];
    __syncthreads();

    int buf = 0;
    for (int kt = 0; kt < D / 32; ++kt) {
        if (kt + 1 < D / 32) {
            *(v8bf*)&Bt[buf ^ 1][row * 32 + cc] =
                *(const v8bf*)&W[(nBase + row) * D + (kt + 1) * 32 + cc];
        }
        v16bf af = load_frag_a(Arow, D, kt * 32, lane);
#pragma unroll
        for (int j = 0; j < 4; ++j) {
            v16bf bf_ = load_frag_b(&Bt[buf][(16 * j) * 32], 32, 0, lane);
            acc[j] = wmma_bf16(af, bf_, acc[j]);
        }
        __syncthreads();
        buf ^= 1;
    }

    if (MODE == 0) {
        const int third = nBase >> 10;          // 0=q 1=k 2=v
        const int h  = (nBase & 1023) >> 6;
        const int b  = m0w >> 11;               // row / S
        const int bh = b * H + h;
        if (third < 2) {
            // RoPE: pair (i, i+32) -> tiles (j, j+2), i = 16*j + nr
#pragma unroll
            for (int v = 0; v < 8; ++v) {
                int s = (m0w + v + 8 * half) & (S - 1);
#pragma unroll
                for (int j = 0; j < 2; ++j) {
                    float fi   = (float)(16 * j + nr);
                    float invf = exp2f(fi * -ROPE_L2);
                    float ang  = (float)s * invf;
                    float sn, cs;
                    __sincosf(ang, &sn, &cs);
                    float x1 = acc[j][v], x2 = acc[j + 2][v];
                    acc[j][v]     = x1 * cs - x2 * sn;
                    acc[j + 2][v] = x1 * sn + x2 * cs;
                }
            }
            __bf16* dst = (third == 0) ? Qo : Ko;
            const float sc = (third == 0) ? 0.125f : 1.0f;  // fold softmax scale into Q
#pragma unroll
            for (int j = 0; j < 4; ++j)
#pragma unroll
                for (int v = 0; v < 8; ++v) {
                    int s = (m0w + v + 8 * half) & (S - 1);
                    dst[(bh * S + s) * HD + 16 * j + nr] = (__bf16)(acc[j][v] * sc);
                }
        } else {
            // V stored transposed: [B][H][HD][S]
#pragma unroll
            for (int j = 0; j < 4; ++j)
#pragma unroll
                for (int v = 0; v < 8; ++v) {
                    int s = (m0w + v + 8 * half) & (S - 1);
                    Vto[(bh * HD + 16 * j + nr) * S + s] = (__bf16)acc[j][v];
                }
        }
    } else {
#pragma unroll
        for (int j = 0; j < 4; ++j)
#pragma unroll
            for (int v = 0; v < 8; ++v) {
                int m = m0w + v + 8 * half;
                int n = nBase + 16 * j + nr;
                outF[m * D + n] = acc[j][v] + bias[n];
            }
    }
}

// ---- Flash attention ------------------------------------------------------
// grid (S/128, B*H); 8 waves/block; wave owns 16 query rows.
// K/V 64-key tiles staged into LDS by the Tensor Data Mover (wave 0 issues,
// double-buffered, TENSORcnt + one barrier per tile), all 8 waves consume.
__global__ void __launch_bounds__(256)
attn_kernel(const __bf16* __restrict__ Q, const __bf16* __restrict__ K,
            const __bf16* __restrict__ Vt, __bf16* __restrict__ Ao) {
    __shared__ __bf16 Ktile[2][64 * 64];    // 16 KB: rows=t, cols=hd
    __shared__ __bf16 Vtile[2][64 * 64];    // 16 KB: rows=hd, cols=t
    __shared__ __bf16 Pbuf[8][16 * 64];     // 16 KB: per-wave P tile
    const int wave = threadIdx.x >> 5;
    const int lane = threadIdx.x & 31;
    const int half = lane >> 4;
    const int nr   = lane & 15;
    const int bh = blockIdx.y;
    const int b  = bh >> 4;
    const int h  = bh & 15;
    const __bf16* Qp = Q + bh * (S * HD);
    const unsigned long long Kg = (unsigned long long)(size_t)(K  + bh * (S * HD));
    const unsigned long long Vg = (unsigned long long)(size_t)(Vt + bh * (HD * S));
    const int qRow0 = blockIdx.x * 128 + wave * 16;

    v16bf qf[2];
    qf[0] = load_frag_a(Qp + qRow0 * HD, HD, 0, lane);
    qf[1] = load_frag_a(Qp + qRow0 * HD, HD, 32, lane);

    v8f accO[4] = {};
    float m_i[8], l_i[8];
#pragma unroll
    for (int v = 0; v < 8; ++v) { m_i[v] = -1e30f; l_i[v] = 0.f; }

    __bf16* pb = &Pbuf[wave][0];

    // prologue: DMA tile 0
    if (wave == 0) {
        tdm_load_2d(lds_off(&Ktile[0][0]), Kg, HD, S, HD, HD, 64);
        tdm_load_2d(lds_off(&Vtile[0][0]), Vg, S, HD, S, 64, HD);
        __builtin_amdgcn_s_wait_tensorcnt(0);
    }
    __syncthreads();

    constexpr int NT = S / 64;
    for (int it = 0; it < NT; ++it) {
        const int buf = it & 1;
        if (wave == 0 && it + 1 < NT) {     // DMA next tile into other buffer
            const int ktn = (it + 1) * 64;
            tdm_load_2d(lds_off(&Ktile[buf ^ 1][0]), Kg + (size_t)ktn * HD * 2, HD, S, HD, HD, 64);
            tdm_load_2d(lds_off(&Vtile[buf ^ 1][0]), Vg + (size_t)ktn * 2,      S, HD, S, 64, HD);
        }
        // scores = Q * K^T  (Q pre-scaled by 1/8); K tile from LDS
        v8f sc[4] = {};
#pragma unroll
        for (int j = 0; j < 4; ++j)
#pragma unroll
            for (int ks = 0; ks < 2; ++ks) {
                v16bf kf = load_frag_b(&Ktile[buf][(16 * j) * 64], 64, 32 * ks, lane);
                sc[j] = wmma_bf16(qf[ks], kf, sc[j]);
            }
        // online softmax; row m = v + 8*half lives in VGPR v across a 16-lane half
#pragma unroll
        for (int v = 0; v < 8; ++v) {
            float mx = fmaxf(fmaxf(sc[0][v], sc[1][v]), fmaxf(sc[2][v], sc[3][v]));
            mx = redmax16(mx);
            float mnew = fmaxf(m_i[v], mx);
            float corr = __expf(m_i[v] - mnew);
            float sum = 0.f;
#pragma unroll
            for (int j = 0; j < 4; ++j) {
                float p = __expf(sc[j][v] - mnew);
                sc[j][v] = p;
                sum += p;
            }
            sum = redsum16(sum);
            l_i[v] = l_i[v] * corr + sum;
            m_i[v] = mnew;
#pragma unroll
            for (int j = 0; j < 4; ++j) accO[j][v] *= corr;
        }
        // C-layout -> A-operand re-layout through per-wave LDS tile
#pragma unroll
        for (int j = 0; j < 4; ++j)
#pragma unroll
            for (int v = 0; v < 8; ++v)
                pb[(v + 8 * half) * 64 + 16 * j + nr] = (__bf16)sc[j][v];
        asm volatile("s_wait_dscnt 0" ::: "memory");   // same-wave LDS RAW fence
        // O += P * V; V tile from LDS (transposed -> contiguous B-operand rows)
#pragma unroll
        for (int ks = 0; ks < 2; ++ks) {
            v16bf pf = load_frag_a(pb, 64, 32 * ks, lane);
#pragma unroll
            for (int j = 0; j < 4; ++j) {
                v16bf vf = load_frag_b(&Vtile[buf][(16 * j) * 64], 64, 32 * ks, lane);
                accO[j] = wmma_bf16(pf, vf, accO[j]);
            }
        }
        if (wave == 0 && it + 1 < NT) __builtin_amdgcn_s_wait_tensorcnt(0);
        __syncthreads();
    }

#pragma unroll
    for (int v = 0; v < 8; ++v) {
        float inv = 1.0f / l_i[v];
        int rowq = qRow0 + v + 8 * half;
#pragma unroll
        for (int j = 0; j < 4; ++j)
            Ao[(b * S + rowq) * D + h * HD + 16 * j + nr] = (__bf16)(accO[j][v] * inv);
    }
}

extern "C" void kernel_launch(void* const* d_in, const int* in_sizes, int n_in,
                              void* d_out, int out_size, void* d_ws, size_t ws_size,
                              hipStream_t stream) {
    const float* x     = (const float*)d_in[0];   // [B,S,D]
    const float* w_qkv = (const float*)d_in[1];   // [3D,D]
    const float* w_out = (const float*)d_in[2];   // [D,D]
    const float* b_out = (const float*)d_in[3];   // [D]

    char* ws = (char*)d_ws;
    __bf16* xb   = (__bf16*)(ws + (size_t)0);            // 8 MB
    __bf16* wqb  = (__bf16*)(ws + ((size_t)8  << 20));   // 6 MB
    __bf16* wob  = (__bf16*)(ws + ((size_t)14 << 20));   // 2 MB
    __bf16* Qb   = (__bf16*)(ws + ((size_t)16 << 20));   // 8 MB  [B][H][S][64]
    __bf16* Kb   = (__bf16*)(ws + ((size_t)24 << 20));   // 8 MB  [B][H][S][64]
    __bf16* Vtb  = (__bf16*)(ws + ((size_t)32 << 20));   // 8 MB  [B][H][64][S]
    __bf16* Attb = (__bf16*)(ws + ((size_t)40 << 20));   // 8 MB  [M][D]

    const int nX = M * D, nWq = 3 * D * D, nWo = D * D;
    cvt_bf16_kernel<<<(nX  + 255) / 256, 256, 0, stream>>>(x,     xb,  nX);
    cvt_bf16_kernel<<<(nWq + 255) / 256, 256, 0, stream>>>(w_qkv, wqb, nWq);
    cvt_bf16_kernel<<<(nWo + 255) / 256, 256, 0, stream>>>(w_out, wob, nWo);

    // QKV projection + RoPE + layout
    gemm_nt_kernel<0><<<dim3(M / 128, (3 * D) / 64), 256, 0, stream>>>(
        xb, wqb, nullptr, nullptr, Qb, Kb, Vtb);

    // Flash attention (TDM-staged K/V tiles)
    attn_kernel<<<dim3(S / 128, Bsz * H), 256, 0, stream>>>(Qb, Kb, Vtb, Attb);

    // Output projection + bias -> fp32 d_out
    gemm_nt_kernel<1><<<dim3(M / 128, D / 64), 256, 0, stream>>>(
        Attb, wob, (float*)d_out, b_out, nullptr, nullptr, nullptr);
}
